// SparseMoeBlock_26998164423121
// MI455X (gfx1250) — matrix operand order repaired and checked by reading, not compiled, not run on hardware
//
#include <hip/hip_runtime.h>

// ---------------------------------------------------------------------------
// SparseMoeBlock for MI455X (gfx1250, wave32, WMMA bf16).
//   1) router_kernel : fused router MLP + softmax + top-2 -> cw [T,E] in ws,
//                      router_logits -> second region of d_out.
//   2) transpose_cvt : fp32 [R,C] -> bf16 [C,R] (K-major weights) in ws.
//   3) moe_expert    : per 16-token block, fused
//                      out = sum_e cw[:,e]*(gelu(h@w1[e]+b1[e])@w2[e]+b2[e])
//                      on v_wmma_f32_16x16x32_bf16; stage-2 B operand is
//                      double-buffered through LDS with
//                      global_load_async_to_lds_b128 (ASYNCcnt) when available.
// Workspace: cw (128 KB) + w1t (64 MB) + w2t (64 MB) ~= 128.1 MiB.
// ---------------------------------------------------------------------------

typedef __bf16 bf16_t;
typedef __attribute__((ext_vector_type(16))) __bf16 v16bf;
typedef __attribute__((ext_vector_type(8)))  __bf16 v8bf;
typedef __attribute__((ext_vector_type(8)))  float  v8f;
typedef __attribute__((ext_vector_type(4)))  int    v4i;

union Frag {
    v16bf v;
    v8bf  h[2];
};

#define T_TOK 4096
#define DDIM  1024
#define EXP   8
#define FDIM  4096
#define FCHUNK 512
#define HPAD  (DDIM + 8)     // bf16 elements per LDS row of token block
#define APAD  (FCHUNK + 8)   // bf16 elements per LDS row of act chunk

// per-wave async B staging: 2 buffers x 8 tiles x 512 bf16 (1KB/tile)
#define BBUF_ELEMS 4096                   // one buffer: 8*512
#define BSTAGE_ELEMS (8 * 2 * BBUF_ELEMS) // 8 waves, double buffered
#define LDS_ELEMS (16 * HPAD + 16 * APAD + BSTAGE_ELEMS)

#if __has_builtin(__builtin_amdgcn_global_load_async_to_lds_b128)
#define HAS_ASYNC 1
#else
#define HAS_ASYNC 0
#endif

#if HAS_ASYNC
typedef __attribute__((address_space(1))) v4i gv4i;   // int4 in global AS
typedef __attribute__((address_space(3))) v4i lv4i;   // int4 in LDS AS
__device__ __forceinline__ void async_cp16(const bf16_t* g, bf16_t* l) {
    // per-lane 16B global -> LDS copy, tracked by ASYNCcnt
    __builtin_amdgcn_global_load_async_to_lds_b128(
        (gv4i*)(unsigned long long)g,   // flat == global bit pattern
        (lv4i*)l,                       // addrspacecast flat -> LDS offset
        0, 0);
}
#endif

// ---------------------------------------------------------------------------
// Router: one block per token.
// ---------------------------------------------------------------------------
__global__ __launch_bounds__(256) void moe_router_kernel(
    const float* __restrict__ h,
    const float* __restrict__ gw1, const float* __restrict__ gb1,
    const float* __restrict__ gw2, const float* __restrict__ gb2,
    const float* __restrict__ gw3,
    float* __restrict__ cw, float* __restrict__ logits_out)
{
    __shared__ float hrow[DDIM];
    __shared__ float g1s[256];
    __shared__ float g2s[64];
    __shared__ float lgs[8];

    const int t   = blockIdx.x;
    const int tid = threadIdx.x;

    #pragma unroll
    for (int i = 0; i < 4; ++i)
        hrow[i * 256 + tid] = h[(size_t)t * DDIM + i * 256 + tid];
    __syncthreads();

    float a1 = 0.0f;
    #pragma unroll 4
    for (int k = 0; k < DDIM; ++k)
        a1 = fmaf(hrow[k], gw1[(size_t)k * 256 + tid], a1);
    g1s[tid] = fmaxf(a1 + gb1[tid], 0.0f);
    __syncthreads();

    if (tid < 64) {
        float a2 = 0.0f;
        #pragma unroll 4
        for (int k = 0; k < 256; ++k)
            a2 = fmaf(g1s[k], gw2[(size_t)k * 64 + tid], a2);
        g2s[tid] = fmaxf(a2 + gb2[tid], 0.0f);
    }
    __syncthreads();

    if (tid < EXP) {
        float a3 = 0.0f;
        #pragma unroll 4
        for (int k = 0; k < 64; ++k)
            a3 = fmaf(g2s[k], gw3[(size_t)k * EXP + tid], a3);
        lgs[tid] = a3;
        logits_out[(size_t)t * EXP + tid] = a3;
    }
    __syncthreads();

    if (tid == 0) {
        float m = lgs[0];
        #pragma unroll
        for (int j = 1; j < EXP; ++j) m = fmaxf(m, lgs[j]);
        float p[EXP], s = 0.0f;
        #pragma unroll
        for (int j = 0; j < EXP; ++j) { p[j] = __expf(lgs[j] - m); s += p[j]; }
        float inv = 1.0f / s;
        #pragma unroll
        for (int j = 0; j < EXP; ++j) p[j] *= inv;
        int i1 = 0;
        #pragma unroll
        for (int j = 1; j < EXP; ++j) if (p[j] > p[i1]) i1 = j;
        int i2 = (i1 == 0) ? 1 : 0;
        #pragma unroll
        for (int j = 0; j < EXP; ++j) if (j != i1 && p[j] > p[i2]) i2 = j;
        float s2 = 1.0f / (p[i1] + p[i2]);
        float* cwrow = cw + (size_t)t * EXP;
        #pragma unroll
        for (int j = 0; j < EXP; ++j) cwrow[j] = 0.0f;
        cwrow[i1] = p[i1] * s2;
        cwrow[i2] = p[i2] * s2;
    }
}

// ---------------------------------------------------------------------------
// fp32 [R,C] row-major -> bf16 [C,R] row-major. grid = (C/32, R/32, E).
// ---------------------------------------------------------------------------
__global__ __launch_bounds__(256) void transpose_cvt_kernel(
    const float* __restrict__ src, bf16_t* __restrict__ dst, int R, int C)
{
    __shared__ float tile[32][33];
    const size_t mat = (size_t)R * C;
    src += (size_t)blockIdx.z * mat;
    dst += (size_t)blockIdx.z * mat;
    const int c0 = blockIdx.x * 32;
    const int r0 = blockIdx.y * 32;
    const int tx = threadIdx.x & 31;
    const int ty = threadIdx.x >> 5;
    #pragma unroll
    for (int i = 0; i < 32; i += 8)
        tile[ty + i][tx] = src[(size_t)(r0 + ty + i) * C + c0 + tx];
    __syncthreads();
    #pragma unroll
    for (int i = 0; i < 32; i += 8)
        dst[(size_t)(c0 + ty + i) * R + r0 + tx] = (bf16_t)tile[tx][ty + i];
}

// ---------------------------------------------------------------------------
// Fused expert compute. One workgroup (8 wave32) owns 16 tokens x all D cols.
// ---------------------------------------------------------------------------
__device__ __forceinline__ float gelu_tanh(float x) {
    const float c = 0.7978845608028654f;   // sqrt(2/pi)
    float u = c * (x + 0.044715f * x * x * x);
    return 0.5f * x * (1.0f + tanhf(u));
}

__global__ __launch_bounds__(256) void moe_expert_kernel(
    const float*  __restrict__ h,    // [T,D] fp32
    const float*  __restrict__ b1,   // [E,F]
    const float*  __restrict__ b2,   // [E,D]
    const float*  __restrict__ cw,   // [T,E] combine weights (ws)
    const bf16_t* __restrict__ w1t,  // [E,F,D] bf16, K(D)-major
    const bf16_t* __restrict__ w2t,  // [E,D,F] bf16, K(F)-major
    float* __restrict__ out)         // [T,D]
{
    extern __shared__ bf16_t smem[];
    bf16_t* hS = smem;                 // 16 x HPAD
    bf16_t* aS = hS + 16 * HPAD;       // 16 x APAD
    bf16_t* bS = aS + 16 * APAD;       // async B staging (per-wave private)

    const int tid  = threadIdx.x;
    const int wave = tid >> 5;
    const int lane = tid & 31;
    const int l16  = lane & 15;
    const int half = lane >> 4;        // 0: lanes 0-15, 1: lanes 16-31
    const int tok0 = blockIdx.x * 16;

    // Stage token block into LDS as bf16 (A operand of stage-1 GEMM).
    for (int idx = tid; idx < 16 * DDIM; idx += 256) {
        int r = idx >> 10, c = idx & (DDIM - 1);
        hS[r * HPAD + c] = (bf16_t)h[(size_t)(tok0 + r) * DDIM + c];
    }
    __syncthreads();

    const int outColBase = wave * 128;          // this wave's 128 output cols
    bf16_t* myB = bS + wave * 2 * BBUF_ELEMS;   // private double buffer

    v8f accS[8];                                // running sum over experts
    #pragma unroll
    for (int t = 0; t < 8; ++t) accS[t] = (v8f){};

    for (int e = 0; e < EXP; ++e) {
        const bf16_t* w1e = w1t + (size_t)e * FDIM * DDIM;
        const bf16_t* w2e = w2t + (size_t)e * DDIM * FDIM;

        // routing weight per row owned by this lane-half (M = r + half*8)
        float wrow[8];
        #pragma unroll
        for (int r = 0; r < 8; ++r)
            wrow[r] = cw[(size_t)(tok0 + r + half * 8) * EXP + e]; // * SCALE(=1)

        for (int fc = 0; fc < FDIM; fc += FCHUNK) {
            // ---- stage 1: acc1 = h(16xD) @ w1[:, fc..fc+FCHUNK) ----
            v8f acc1[4];
            #pragma unroll
            for (int t = 0; t < 4; ++t) acc1[t] = (v8f){};

            for (int k0 = 0; k0 < DDIM; k0 += 32) {
                Frag a;   // A 16x32 bf16: lane row = l16, 16 contig K vals
                const bf16_t* ap = &hS[l16 * HPAD + k0 + half * 16];
                a.h[0] = *(const v8bf*)(ap);
                a.h[1] = *(const v8bf*)(ap + 8);
                #pragma unroll
                for (int t = 0; t < 4; ++t) {
                    const int ncol = fc + wave * 64 + t * 16 + l16;   // F index
                    Frag b;   // B 32x16 bf16: lane col = l16, 16 contig K vals
                    b.v = *(const v16bf*)(w1e + (size_t)ncol * DDIM + k0 + half * 16);
                    acc1[t] = __builtin_amdgcn_wmma_f32_16x16x32_bf16(
                        false, a.v, false, b.v, (short)0, acc1[t], false, false);
                    if (t == 0 && k0 + 32 < DDIM)
                        __builtin_prefetch(w1e + (size_t)ncol * DDIM + k0 + 32, 0, 1);
                }
            }
            // bias + gelu -> bf16 activation chunk in LDS
            #pragma unroll
            for (int t = 0; t < 4; ++t) {
                const int colLocal = wave * 64 + t * 16 + l16;
                const float bias = b1[(size_t)e * FDIM + fc + colLocal];
                #pragma unroll
                for (int r = 0; r < 8; ++r) {
                    float g = gelu_tanh(acc1[t][r] + bias);
                    aS[(r + half * 8) * APAD + colLocal] = (bf16_t)g;
                }
            }
            __syncthreads();

            // ---- stage 2: accC = act(16xFCHUNK) @ w2[fc..fc+FCHUNK, cols) ----
            v8f accC[8];
            #pragma unroll
            for (int t = 0; t < 8; ++t) accC[t] = (v8f){};

#if HAS_ASYNC
            // per-lane base of this wave's B fragment stream for this chunk
            const bf16_t* gbase =
                w2e + (size_t)(outColBase + l16) * FDIM + fc + half * 16;
            // prologue: queue k-step 0 into buffer 0 (16 async 16B copies)
            {
                bf16_t* dst = myB + lane * 16;
                #pragma unroll
                for (int t = 0; t < 8; ++t) {
                    const bf16_t* g = gbase + (size_t)t * 16 * FDIM;
                    async_cp16(g,     dst + t * 512);
                    async_cp16(g + 8, dst + t * 512 + 8);
                }
            }
            for (int kk = 0; kk < FCHUNK; kk += 32) {
                const int cur = (kk >> 5) & 1;
                bf16_t* curB = myB + cur * BBUF_ELEMS;
                if (kk + 32 < FCHUNK) {
                    // queue next k-step into the other buffer
                    bf16_t* dst = myB + (cur ^ 1) * BBUF_ELEMS + lane * 16;
                    #pragma unroll
                    for (int t = 0; t < 8; ++t) {
                        const bf16_t* g = gbase + (size_t)t * 16 * FDIM + kk + 32;
                        async_cp16(g,     dst + t * 512);
                        async_cp16(g + 8, dst + t * 512 + 8);
                    }
                    // current buffer's 16 copies (issued earlier) are done
                    asm volatile("s_wait_asynccnt 0x10" ::: "memory");
                } else {
                    asm volatile("s_wait_asynccnt 0x0" ::: "memory");
                }
                Frag a;
                const bf16_t* ap = &aS[l16 * APAD + kk + half * 16];
                a.h[0] = *(const v8bf*)(ap);
                a.h[1] = *(const v8bf*)(ap + 8);
                #pragma unroll
                for (int t = 0; t < 8; ++t) {
                    Frag b;   // read back this lane's staged 32B fragment
                    const bf16_t* lp = curB + t * 512 + lane * 16;
                    b.h[0] = *(const v8bf*)(lp);
                    b.h[1] = *(const v8bf*)(lp + 8);
                    accC[t] = __builtin_amdgcn_wmma_f32_16x16x32_bf16(
                        false, a.v, false, b.v, (short)0, accC[t], false, false);
                }
            }
#else
            for (int kk = 0; kk < FCHUNK; kk += 32) {
                Frag a;
                const bf16_t* ap = &aS[l16 * APAD + kk + half * 16];
                a.h[0] = *(const v8bf*)(ap);
                a.h[1] = *(const v8bf*)(ap + 8);
                #pragma unroll
                for (int t = 0; t < 8; ++t) {
                    const int ncol = outColBase + t * 16 + l16;        // D index
                    Frag b;
                    b.v = *(const v16bf*)(w2e + (size_t)ncol * FDIM + fc + kk + half * 16);
                    accC[t] = __builtin_amdgcn_wmma_f32_16x16x32_bf16(
                        false, a.v, false, b.v, (short)0, accC[t], false, false);
                    if (t == 0 && kk + 32 < FCHUNK)
                        __builtin_prefetch(w2e + (size_t)ncol * FDIM + fc + kk + 32, 0, 1);
                }
            }
#endif
            // fold this chunk into the expert-weighted running sum
            #pragma unroll
            for (int t = 0; t < 8; ++t)
                #pragma unroll
                for (int r = 0; r < 8; ++r)
                    accS[t][r] = fmaf(accC[t][r], wrow[r], accS[t][r]);
            __syncthreads();   // aS reused next chunk
        }

        // expert bias, weighted by the same routing weight
        #pragma unroll
        for (int t = 0; t < 8; ++t) {
            const float bias2 = b2[(size_t)e * DDIM + outColBase + t * 16 + l16];
            #pragma unroll
            for (int r = 0; r < 8; ++r)
                accS[t][r] = fmaf(bias2, wrow[r], accS[t][r]);
        }
    }

    // Write final rows (exclusive to this workgroup -> plain stores).
    #pragma unroll
    for (int t = 0; t < 8; ++t) {
        const int col = outColBase + t * 16 + l16;
        #pragma unroll
        for (int r = 0; r < 8; ++r) {
            const int row = tok0 + r + half * 8;
            out[(size_t)row * DDIM + col] = accS[t][r];
        }
    }
}

// ---------------------------------------------------------------------------
extern "C" void kernel_launch(void* const* d_in, const int* in_sizes, int n_in,
                              void* d_out, int out_size, void* d_ws, size_t ws_size,
                              hipStream_t stream)
{
    const float* h   = (const float*)d_in[0];
    const float* gw1 = (const float*)d_in[1];
    const float* gb1 = (const float*)d_in[2];
    const float* gw2 = (const float*)d_in[3];
    const float* gb2 = (const float*)d_in[4];
    const float* gw3 = (const float*)d_in[5];
    const float* w1  = (const float*)d_in[6];
    const float* b1  = (const float*)d_in[7];
    const float* w2  = (const float*)d_in[8];
    const float* b2  = (const float*)d_in[9];

    float* out    = (float*)d_out;                       // [T,D]
    float* logits = out + (size_t)T_TOK * DDIM;          // [T,E]

    char* ws = (char*)d_ws;
    float*  cw  = (float*)ws;                                          // T*E f32
    bf16_t* w1t = (bf16_t*)(ws + (size_t)T_TOK * EXP * sizeof(float)); // E*F*D
    bf16_t* w2t = w1t + (size_t)EXP * FDIM * DDIM;                     // E*D*F

    moe_router_kernel<<<T_TOK, 256, 0, stream>>>(h, gw1, gb1, gw2, gb2, gw3,
                                                 cw, logits);

    // w1 [E, D, F] -> w1t [E, F, D]
    transpose_cvt_kernel<<<dim3(FDIM / 32, DDIM / 32, EXP), 256, 0, stream>>>(
        w1, w1t, DDIM, FDIM);
    // w2 [E, F, D] -> w2t [E, D, F]
    transpose_cvt_kernel<<<dim3(DDIM / 32, FDIM / 32, EXP), 256, 0, stream>>>(
        w2, w2t, FDIM, DDIM);

    const size_t lds_bytes = (size_t)LDS_ELEMS * sizeof(bf16_t);
    moe_expert_kernel<<<T_TOK / 16, 256, lds_bytes, stream>>>(h, b1, b2, cw,
                                                              w1t, w2t, out);
}
